// TransformerBlock_15264313770254
// MI455X (gfx1250) — matrix operand order repaired and checked by reading, not compiled, not run on hardware
//
#include <hip/hip_runtime.h>
#include <math.h>

// ---- problem constants (from reference) ----
#define B_  2
#define S_  2048
#define E_  1536
#define H_  16
#define D_  96
#define HID_ 6144
#define BS_ (B_ * S_)   // 4096 rows

typedef __attribute__((ext_vector_type(16))) __bf16 v16bf;
typedef __attribute__((ext_vector_type(8)))  float  v8f;

static __device__ __forceinline__ v8f v8f_zero() {
    v8f z;
#pragma unroll
    for (int i = 0; i < 8; ++i) z[i] = 0.0f;
    return z;
}

static __device__ __forceinline__ v8f wmma_bf16(v16bf a, v16bf b, v8f c) {
    // D = A(16x32 bf16) * B(32x16 bf16) + C(16x16 f32)
    return __builtin_amdgcn_wmma_f32_16x16x32_bf16(
        /*neg_a=*/false, a, /*neg_b=*/false, b,
        /*c_mod=*/(short)0, c, /*reuse_a=*/false, /*reuse_b=*/false);
}

// ---------------------------------------------------------------------------
// fp32 -> bf16 conversion (weights & activations)
// ---------------------------------------------------------------------------
__global__ __launch_bounds__(256) void cvt_f32_bf16_kernel(
    const float* __restrict__ src, __bf16* __restrict__ dst, size_t n) {
    for (size_t i = (size_t)blockIdx.x * 256 + threadIdx.x; i < n;
         i += (size_t)gridDim.x * 256)
        dst[i] = (__bf16)src[i];
}

// ---------------------------------------------------------------------------
// LayerNorm (one block per row) -> bf16 output
// ---------------------------------------------------------------------------
__global__ __launch_bounds__(256) void ln_bf16_kernel(
    const float* __restrict__ x, const float* __restrict__ g,
    const float* __restrict__ b, __bf16* __restrict__ out, int E) {
    const int row = blockIdx.x;
    const float* xr = x + (size_t)row * E;
    float s = 0.f, s2 = 0.f;
    for (int c = threadIdx.x; c < E; c += 256) {
        float v = xr[c];
        s += v;
        s2 += v * v;
    }
    __shared__ float sh1[256], sh2[256];
    sh1[threadIdx.x] = s;
    sh2[threadIdx.x] = s2;
    __syncthreads();
    for (int off = 128; off > 0; off >>= 1) {
        if (threadIdx.x < off) {
            sh1[threadIdx.x] += sh1[threadIdx.x + off];
            sh2[threadIdx.x] += sh2[threadIdx.x + off];
        }
        __syncthreads();
    }
    const float mu  = sh1[0] / (float)E;
    const float var = sh2[0] / (float)E - mu * mu;
    const float inv = rsqrtf(var + 1e-5f);
    for (int c = threadIdx.x; c < E; c += 256)
        out[(size_t)row * E + c] = (__bf16)((xr[c] - mu) * inv * g[c] + b[c]);
}

// ---------------------------------------------------------------------------
// Tiled WMMA GEMM: C[M,N] = A_bf16[M,K] @ W_bf16[K,ldw](cols) + epilogue
//   MODE 0: out_bf16 = acc + bias
//   MODE 1: out_f32  = acc + bias + resid
//   MODE 2: GEGLU: out_bf16 = gelu(acc_a + b1[col]) * (acc_g + b1[col+N])
//           (W has 2N columns, ldw = 2N; second accumulator set for gate)
// Block tile 128x64, K-slab 32, 256 threads = 8 waves, each wave: 16 rows x 64 cols.
// ---------------------------------------------------------------------------
template <int MODE>
__global__ __launch_bounds__(256) void wmma_gemm_kernel(
    const __bf16* __restrict__ A, const __bf16* __restrict__ W,
    const float* __restrict__ bias, const float* __restrict__ resid,
    void* __restrict__ outv, int M, int N, int K, int ldw) {
    constexpr int BM = 128, BN = 64, BK = 32, BKP = 40;  // pad LDS rows
    __shared__ __bf16 As[BM][BKP];
    __shared__ __bf16 Bs[(MODE == 2 ? 2 : 1) * BN][BKP];

    const int tid   = threadIdx.x;
    const int wave  = tid >> 5;
    const int lane  = tid & 31;
    const int ln16  = lane & 15;
    const int khalf = lane >> 4;
    const int m0 = blockIdx.y * BM;
    const int n0 = blockIdx.x * BN;

    v8f acc[4], acc2[4];
#pragma unroll
    for (int j = 0; j < 4; ++j) { acc[j] = v8f_zero(); acc2[j] = v8f_zero(); }

    for (int k0 = 0; k0 < K; k0 += BK) {
        __syncthreads();  // protect previous slab's fragment reads
        // --- stage A tile (row-major, 16B vector loads) ---
        for (int i = tid; i < BM * (BK / 8); i += 256) {
            const int r = i >> 2, seg = i & 3;
            const __bf16* gp = A + (size_t)(m0 + r) * K + k0 + seg * 8;
            *reinterpret_cast<uint4*>(&As[r][seg * 8]) =
                *reinterpret_cast<const uint4*>(gp);
            if (k0 + BK < K) __builtin_prefetch(gp + BK, 0, 1);
        }
        // --- stage B tile transposed -> Bs[n][k] (coalesced over n) ---
        for (int i = tid; i < BN * BK; i += 256) {
            const int n = i & (BN - 1), kk = i >> 6;
            const int col = n0 + n;
            __bf16 val = (__bf16)0.0f;
            if (MODE == 2 || col < N) val = W[(size_t)(k0 + kk) * ldw + col];
            Bs[n][kk] = val;
            if constexpr (MODE == 2)
                Bs[BN + n][kk] = W[(size_t)(k0 + kk) * ldw + col + N];
        }
        __syncthreads();

        // --- A fragment: lane holds row (lane&15), K = 16*khalf + i
        v16bf af;
#pragma unroll
        for (int i = 0; i < 8; ++i) {
            af[i]     = As[wave * 16 + ln16][khalf * 16 + i];
            af[8 + i] = As[wave * 16 + ln16][khalf * 16 + 8 + i];
        }
#pragma unroll
        for (int j = 0; j < 4; ++j) {
            // B fragment: lane holds col (lane&15), K = 16*khalf + i
            v16bf bf;
#pragma unroll
            for (int i = 0; i < 16; ++i) bf[i] = Bs[j * 16 + ln16][khalf * 16 + i];
            acc[j] = wmma_bf16(af, bf, acc[j]);
            if constexpr (MODE == 2) {
                v16bf bg;
#pragma unroll
                for (int i = 0; i < 16; ++i)
                    bg[i] = Bs[BN + j * 16 + ln16][khalf * 16 + i];
                acc2[j] = wmma_bf16(af, bg, acc2[j]);
            }
        }
    }

    // --- epilogue: C element (row = r + 8*khalf, col = lane&15) per VGPR r ---
#pragma unroll
    for (int j = 0; j < 4; ++j) {
#pragma unroll
        for (int r = 0; r < 8; ++r) {
            const int row = m0 + wave * 16 + r + khalf * 8;
            const int col = n0 + j * 16 + ln16;
            if (MODE != 2 && col >= N) continue;
            if constexpr (MODE == 0) {
                ((__bf16*)outv)[(size_t)row * N + col] =
                    (__bf16)(acc[j][r] + bias[col]);
            } else if constexpr (MODE == 1) {
                ((float*)outv)[(size_t)row * N + col] =
                    acc[j][r] + bias[col] + resid[(size_t)row * N + col];
            } else {
                const float av = acc[j][r] + bias[col];
                const float gv = acc2[j][r] + bias[col + N];
                const float ge = 0.5f * av * (1.0f + erff(av * 0.70710678118654752f));
                ((__bf16*)outv)[(size_t)row * N + col] = (__bf16)(ge * gv);
            }
        }
    }
}

// ---------------------------------------------------------------------------
// Flash attention (MQA: single K/V head of D=96, non-causal).
// Block = 128 threads (4 waves), 64 Q rows per block, 32 keys per step.
// scores = Q@K^T via WMMA; online softmax with shfl_xor row reductions
// matched to the WMMA C layout; P staged in per-wave LDS to become an
// A-fragment; O += P@V via 6 WMMA N-tiles over D=96.
// ---------------------------------------------------------------------------
__global__ __launch_bounds__(128) void attn_kernel(
    const __bf16* __restrict__ q, const __bf16* __restrict__ k,
    const __bf16* __restrict__ v, __bf16* __restrict__ attn) {
    __shared__ __bf16 Qs[64][104];
    __shared__ __bf16 Ks[32][104];
    __shared__ __bf16 Vs[32][104];
    __shared__ __bf16 Ps[4][16][32];

    const int tid = threadIdx.x, wave = tid >> 5, lane = tid & 31;
    const int ln16 = lane & 15, khalf = lane >> 4;
    const int s0 = blockIdx.x * 64;
    const int bh = blockIdx.y, b = bh / H_, h = bh % H_;
    const float scale = 0.1020620726159658f;  // 1/sqrt(96)

    // stage Q tile (64 x 96)
    for (int i = tid; i < 64 * 12; i += 128) {
        const int r = i / 12, seg = i % 12;
        *reinterpret_cast<uint4*>(&Qs[r][seg * 8]) =
            *reinterpret_cast<const uint4*>(
                q + (size_t)(b * S_ + s0 + r) * E_ + h * D_ + seg * 8);
    }

    float mrun[8], lrun[8], alpha[8];
#pragma unroll
    for (int r = 0; r < 8; ++r) { mrun[r] = -1e30f; lrun[r] = 0.0f; }
    v8f oacc[6];
#pragma unroll
    for (int j = 0; j < 6; ++j) oacc[j] = v8f_zero();

    for (int t0 = 0; t0 < S_; t0 += 32) {
        __syncthreads();
        for (int i = tid; i < 32 * 12; i += 128) {
            const int r = i / 12, seg = i % 12;
            *reinterpret_cast<uint4*>(&Ks[r][seg * 8]) =
                *reinterpret_cast<const uint4*>(
                    k + (size_t)(b * S_ + t0 + r) * D_ + seg * 8);
            *reinterpret_cast<uint4*>(&Vs[r][seg * 8]) =
                *reinterpret_cast<const uint4*>(
                    v + (size_t)(b * S_ + t0 + r) * D_ + seg * 8);
        }
        __syncthreads();

        // scores: 16 rows x 32 keys, K-dim = 96 (3 slabs)
        v8f sa0 = v8f_zero(), sa1 = v8f_zero();
#pragma unroll
        for (int dsl = 0; dsl < 96; dsl += 32) {
            v16bf af;
#pragma unroll
            for (int i = 0; i < 8; ++i) {
                af[i]     = Qs[wave * 16 + ln16][dsl + khalf * 16 + i];
                af[8 + i] = Qs[wave * 16 + ln16][dsl + khalf * 16 + 8 + i];
            }
            v16bf b0, b1v;
#pragma unroll
            for (int i = 0; i < 16; ++i) {
                b0[i]  = Ks[ln16][dsl + khalf * 16 + i];       // keys 0..15
                b1v[i] = Ks[16 + ln16][dsl + khalf * 16 + i];  // keys 16..31
            }
            sa0 = wmma_bf16(af, b0, sa0);
            sa1 = wmma_bf16(af, b1v, sa1);
        }

        // online softmax; C element (row r + 8*khalf, col ln16 / 16+ln16)
#pragma unroll
        for (int r = 0; r < 8; ++r) {
            const float x0 = sa0[r] * scale, x1 = sa1[r] * scale;
            float mx = fmaxf(x0, x1);
            mx = fmaxf(mx, __shfl_xor(mx, 8, 32));
            mx = fmaxf(mx, __shfl_xor(mx, 4, 32));
            mx = fmaxf(mx, __shfl_xor(mx, 2, 32));
            mx = fmaxf(mx, __shfl_xor(mx, 1, 32));
            const float mnew = fmaxf(mrun[r], mx);
            const float p0 = __expf(x0 - mnew);
            const float p1 = __expf(x1 - mnew);
            const float al = __expf(mrun[r] - mnew);
            mrun[r] = mnew;
            float rs = p0 + p1;
            rs += __shfl_xor(rs, 8, 32);
            rs += __shfl_xor(rs, 4, 32);
            rs += __shfl_xor(rs, 2, 32);
            rs += __shfl_xor(rs, 1, 32);
            lrun[r] = lrun[r] * al + rs;
            alpha[r] = al;
            Ps[wave][r + khalf * 8][ln16]      = (__bf16)p0;
            Ps[wave][r + khalf * 8][16 + ln16] = (__bf16)p1;
        }
#pragma unroll
        for (int j = 0; j < 6; ++j)
#pragma unroll
            for (int r = 0; r < 8; ++r) oacc[j][r] *= alpha[r];

        // P (16x32) as A-fragment (same-wave LDS RAW is in-order)
        v16bf pf;
#pragma unroll
        for (int i = 0; i < 8; ++i) {
            pf[i]     = Ps[wave][ln16][khalf * 16 + i];
            pf[8 + i] = Ps[wave][ln16][khalf * 16 + 8 + i];
        }
#pragma unroll
        for (int j = 0; j < 6; ++j) {
            v16bf vf;
#pragma unroll
            for (int i = 0; i < 16; ++i)
                vf[i] = Vs[khalf * 16 + i][j * 16 + ln16];
            oacc[j] = wmma_bf16(pf, vf, oacc[j]);
        }
    }

#pragma unroll
    for (int j = 0; j < 6; ++j)
#pragma unroll
        for (int r = 0; r < 8; ++r) {
            const int row = s0 + wave * 16 + r + khalf * 8;
            const float o = oacc[j][r] / lrun[r];
            attn[(size_t)(b * S_ + row) * E_ + h * D_ + j * 16 + ln16] = (__bf16)o;
        }
}

// ---------------------------------------------------------------------------
extern "C" void kernel_launch(void* const* d_in, const int* in_sizes, int n_in,
                              void* d_out, int out_size, void* d_ws, size_t ws_size,
                              hipStream_t stream) {
    const float* x     = (const float*)d_in[0];
    const float* ln1_g = (const float*)d_in[1];
    const float* ln1_b = (const float*)d_in[2];
    const float* wq    = (const float*)d_in[3];
    const float* bq    = (const float*)d_in[4];
    const float* wk    = (const float*)d_in[5];
    const float* bk    = (const float*)d_in[6];
    const float* wv    = (const float*)d_in[7];
    const float* bv    = (const float*)d_in[8];
    const float* wo    = (const float*)d_in[9];
    const float* bo    = (const float*)d_in[10];
    const float* ln2_g = (const float*)d_in[11];
    const float* ln2_b = (const float*)d_in[12];
    const float* w1    = (const float*)d_in[13];
    const float* b1    = (const float*)d_in[14];
    const float* w2    = (const float*)d_in[15];
    const float* b2    = (const float*)d_in[16];

    char* wsp = (char*)d_ws;
    size_t off = 0;
    auto carve = [&](size_t bytes) -> void* {
        void* r = wsp + off;
        off = (off + bytes + 255) & ~(size_t)255;
        return r;
    };
    // ~195 MB total workspace
    __bf16* h1   = (__bf16*)carve((size_t)BS_ * E_ * 2);
    __bf16* wq_b = (__bf16*)carve((size_t)E_ * E_ * 2);
    __bf16* wk_b = (__bf16*)carve((size_t)E_ * D_ * 2);
    __bf16* wv_b = (__bf16*)carve((size_t)E_ * D_ * 2);
    __bf16* wo_b = (__bf16*)carve((size_t)E_ * E_ * 2);
    __bf16* w1_b = (__bf16*)carve((size_t)E_ * 2 * HID_ * 2);
    __bf16* w2_b = (__bf16*)carve((size_t)HID_ * E_ * 2);
    __bf16* qb   = (__bf16*)carve((size_t)BS_ * E_ * 2);
    __bf16* kb   = (__bf16*)carve((size_t)BS_ * D_ * 2);
    __bf16* vb   = (__bf16*)carve((size_t)BS_ * D_ * 2);
    __bf16* attn = (__bf16*)carve((size_t)BS_ * E_ * 2);
    float*  x1   = (float*)carve((size_t)BS_ * E_ * 4);
    __bf16* h2   = (__bf16*)carve((size_t)BS_ * E_ * 2);
    __bf16* ff   = (__bf16*)carve((size_t)BS_ * HID_ * 2);

    auto cvt = [&](const float* s, __bf16* d, size_t n) {
        int blocks = (int)((n + 255) / 256);
        if (blocks > 65535) blocks = 65535;
        cvt_f32_bf16_kernel<<<blocks, 256, 0, stream>>>(s, d, n);
    };
    cvt(wq, wq_b, (size_t)E_ * E_);
    cvt(wk, wk_b, (size_t)E_ * D_);
    cvt(wv, wv_b, (size_t)E_ * D_);
    cvt(wo, wo_b, (size_t)E_ * E_);
    cvt(w1, w1_b, (size_t)E_ * 2 * HID_);
    cvt(w2, w2_b, (size_t)HID_ * E_);

    // LN1 -> h1 (bf16)
    ln_bf16_kernel<<<BS_, 256, 0, stream>>>(x, ln1_g, ln1_b, h1, E_);

    // Q/K/V projections (WMMA)
    dim3 gq(E_ / 64, BS_ / 128);
    wmma_gemm_kernel<0><<<gq, 256, 0, stream>>>(h1, wq_b, bq, nullptr, qb,
                                                BS_, E_, E_, E_);
    dim3 gkv((D_ + 63) / 64, BS_ / 128);
    wmma_gemm_kernel<0><<<gkv, 256, 0, stream>>>(h1, wk_b, bk, nullptr, kb,
                                                 BS_, D_, E_, D_);
    wmma_gemm_kernel<0><<<gkv, 256, 0, stream>>>(h1, wv_b, bv, nullptr, vb,
                                                 BS_, D_, E_, D_);

    // Flash attention (WMMA), MQA
    dim3 ga(S_ / 64, B_ * H_);
    attn_kernel<<<ga, 128, 0, stream>>>(qb, kb, vb, attn);

    // attn @ wo + bo + x -> x1 (f32)
    wmma_gemm_kernel<1><<<gq, 256, 0, stream>>>(attn, wo_b, bo, x, x1,
                                                BS_, E_, E_, E_);

    // LN2 -> h2 (bf16)
    ln_bf16_kernel<<<BS_, 256, 0, stream>>>(x1, ln2_g, ln2_b, h2, E_);

    // fused GEGLU: h2 @ w1 (+b1), gelu(a)*gate -> ff (bf16)
    dim3 gm(HID_ / 64, BS_ / 128);
    wmma_gemm_kernel<2><<<gm, 256, 0, stream>>>(h2, w1_b, b1, nullptr, ff,
                                                BS_, HID_, E_, 2 * HID_);

    // ff @ w2 + b2 + x1 -> out (f32)
    wmma_gemm_kernel<1><<<gq, 256, 0, stream>>>(ff, w2_b, b2, x1, (float*)d_out,
                                                BS_, E_, HID_, E_);
}
